// NearestInterpolator_Torch_28011776704993
// MI455X (gfx1250) — matrix operand order repaired and checked by reading, not compiled, not run on hardware
//
#include <hip/hip_runtime.h>
#include <hip/hip_bf16.h>

// Problem constants (from reference): N=16384 points, D=128, Q=16384 queries, V=64
#define N_PTS 16384
#define DIMS  128
#define NQ    16384
#define NV    64
#define NSPLIT 4   // split of the N axis across workgroups

typedef __bf16 bf16_t;
typedef __attribute__((ext_vector_type(16))) __bf16 v16bf;
typedef __attribute__((ext_vector_type(8)))  __bf16 v8bf;
typedef __attribute__((ext_vector_type(8)))  float  v8f;

// ---------------------------------------------------------------------------
// Kernel 1: convert points fp32 -> bf16 hi/lo split, and compute p_sq in fp32.
// One block (128 threads) per point row.
// ---------------------------------------------------------------------------
__global__ void prep_points_kernel(const float* __restrict__ pts,
                                   bf16_t* __restrict__ phi,
                                   bf16_t* __restrict__ plo,
                                   float* __restrict__ psq) {
    __shared__ float red[128];
    const int r = blockIdx.x;
    const int t = threadIdx.x;
    float x = pts[(size_t)r * DIMS + t];
    bf16_t h = (bf16_t)x;                  // RNE truncation to bf16
    bf16_t l = (bf16_t)(x - (float)h);     // residual
    phi[(size_t)r * DIMS + t] = h;
    plo[(size_t)r * DIMS + t] = l;
    red[t] = x * x;
    __syncthreads();
    for (int s = 64; s > 0; s >>= 1) {
        if (t < s) red[t] += red[t + s];
        __syncthreads();
    }
    if (t == 0) psq[r] = red[0];
}

// ---------------------------------------------------------------------------
// Kernel 2: WMMA bf16 split-GEMM + running argmin of (p_sq - 2*dot).
// Block = 256 threads = 8 waves; wave w owns query rows [blockIdx.x*128+w*16, +16).
// blockIdx.y selects a contiguous N/NSPLIT slice of points.
// ---------------------------------------------------------------------------
__global__ __launch_bounds__(256) void nn_argmin_kernel(
    const float*  __restrict__ q,     // [Q, D] fp32 queries
    const bf16_t* __restrict__ phi,   // [N, D] bf16 hi
    const bf16_t* __restrict__ plo,   // [N, D] bf16 lo
    const float*  __restrict__ psq,   // [N] fp32 |p|^2
    float* __restrict__ pval,         // [Q * NSPLIT] partial best score
    int*   __restrict__ pidx)         // [Q * NSPLIT] partial best index
{
    const int lane  = threadIdx.x & 31;
    const int wave  = threadIdx.x >> 5;
    const int row   = lane & 15;     // A: query row in tile / B: point col in tile
    const int half  = lane >> 4;     // K-half selector per ISA layout
    const int qbase = blockIdx.x * 128 + wave * 16;
    const int split = blockIdx.y;
    const int n0     = split * (N_PTS / NSPLIT);
    const int ntiles = (N_PTS / NSPLIT) / 16;

    // ---- Build A (16x128 queries, hi/lo bf16 split) in registers.
    // 16-bit A 16x32 layout: lanes 0-15 hold M=row, elements 0..7 = K k0..k0+7,
    // elements 8..15 = K k0+16..k0+23 with k0 = c*32 + half*8.
    v16bf ahi[4], alo[4];
    const float* qrow = q + (size_t)(qbase + row) * DIMS;
    #pragma unroll
    for (int c = 0; c < 4; ++c) {
        const int k0 = c * 32 + half * 8;
        #pragma unroll
        for (int e = 0; e < 8; ++e) {
            float x0 = qrow[k0 + e];
            float x1 = qrow[k0 + 16 + e];
            bf16_t h0 = (bf16_t)x0;
            bf16_t h1 = (bf16_t)x1;
            ahi[c][e]     = h0;
            ahi[c][8 + e] = h1;
            alo[c][e]     = (bf16_t)(x0 - (float)h0);
            alo[c][8 + e] = (bf16_t)(x1 - (float)h1);
        }
    }

    float bestv[8];
    int   besti[8];
    #pragma unroll
    for (int i = 0; i < 8; ++i) { bestv[i] = 3.4e38f; besti[i] = 0; }

    for (int t = 0; t < ntiles; ++t) {
        const int n = n0 + t * 16 + row;   // this lane's point (B column)
        // B 32x16 layout: lane = column n, elements 0..7 = K c*32 + half*8 + e,
        // elements 8..15 = K c*32 + half*8 + 16 + e (same order as A).
        const bf16_t* prow_hi = phi + (size_t)n * DIMS + half * 8;
        const bf16_t* prow_lo = plo + (size_t)n * DIMS + half * 8;
        if (t + 1 < ntiles) {  // pull next tile toward the WGP (global_prefetch_b8)
            __builtin_prefetch(prow_hi + 16 * DIMS, 0, 3);
            __builtin_prefetch(prow_lo + 16 * DIMS, 0, 3);
        }
        v8f acc0 = {};
        v8f acc1 = {};
        #pragma unroll
        for (int c = 0; c < 4; ++c) {
            v8bf bh0 = *(const v8bf*)(prow_hi + c * 32);
            v8bf bh1 = *(const v8bf*)(prow_hi + c * 32 + 16);
            v8bf bl0 = *(const v8bf*)(prow_lo + c * 32);
            v8bf bl1 = *(const v8bf*)(prow_lo + c * 32 + 16);
            v16bf bh = __builtin_shufflevector(bh0, bh1, 0,1,2,3,4,5,6,7,8,9,10,11,12,13,14,15);
            v16bf bl = __builtin_shufflevector(bl0, bl1, 0,1,2,3,4,5,6,7,8,9,10,11,12,13,14,15);
            // dot ~= Ahi*Bhi + Alo*Bhi + Ahi*Blo  (lo*lo term negligible)
            acc0 = __builtin_amdgcn_wmma_f32_16x16x32_bf16(false, ahi[c], false, bh,
                                                           (short)0, acc0, false, false);
            acc1 = __builtin_amdgcn_wmma_f32_16x16x32_bf16(false, alo[c], false, bh,
                                                           (short)0, acc1, false, false);
            acc0 = __builtin_amdgcn_wmma_f32_16x16x32_bf16(false, ahi[c], false, bl,
                                                           (short)0, acc0, false, false);
        }
        const float pn = psq[n];
        // C layout: VGPR i, lanes 0-15 -> M=i; lanes 16-31 -> M=i+8 (same N=row).
        #pragma unroll
        for (int i = 0; i < 8; ++i) {
            float dot = acc0[i] + acc1[i];
            float s = fmaf(-2.0f, dot, pn);   // p_sq - 2*q.p  (q_sq dropped: constant)
            if (s < bestv[i]) { bestv[i] = s; besti[i] = n; }
        }
    }

    // Cross-lane argmin within each 16-lane half (xor of bits 0..3 stays in-half).
    #pragma unroll
    for (int m = 8; m >= 1; m >>= 1) {
        #pragma unroll
        for (int i = 0; i < 8; ++i) {
            float ov = __shfl_xor(bestv[i], m, 32);
            int   oi = __shfl_xor(besti[i], m, 32);
            if (ov < bestv[i] || (ov == bestv[i] && oi < besti[i])) {
                bestv[i] = ov; besti[i] = oi;
            }
        }
    }
    if (row == 0) {   // lanes 0 and 16 hold results for rows half*8 + i
        #pragma unroll
        for (int i = 0; i < 8; ++i) {
            const int qq = qbase + half * 8 + i;
            pval[(size_t)qq * NSPLIT + split] = bestv[i];
            pidx[(size_t)qq * NSPLIT + split] = besti[i];
        }
    }
}

// ---------------------------------------------------------------------------
// Kernel 3: reduce NSPLIT partials per query, gather values[idx] -> out.
// One block (NV=64 threads) per query.
// ---------------------------------------------------------------------------
__global__ void finalize_kernel(const float* __restrict__ pval,
                                const int*   __restrict__ pidx,
                                const float* __restrict__ values,
                                float* __restrict__ out) {
    __shared__ int sidx;
    const int qq = blockIdx.x;
    const int t  = threadIdx.x;
    if (t == 0) {
        float bv = pval[(size_t)qq * NSPLIT];
        int   bi = pidx[(size_t)qq * NSPLIT];
        for (int s = 1; s < NSPLIT; ++s) {
            float v = pval[(size_t)qq * NSPLIT + s];
            int   ii = pidx[(size_t)qq * NSPLIT + s];
            if (v < bv || (v == bv && ii < bi)) { bv = v; bi = ii; }
        }
        sidx = bi;
    }
    __syncthreads();
    out[(size_t)qq * NV + t] = values[(size_t)sidx * NV + t];
}

// ---------------------------------------------------------------------------
extern "C" void kernel_launch(void* const* d_in, const int* in_sizes, int n_in,
                              void* d_out, int out_size, void* d_ws, size_t ws_size,
                              hipStream_t stream) {
    (void)in_sizes; (void)n_in; (void)out_size; (void)ws_size;
    const float* points  = (const float*)d_in[0];  // [N, D]
    const float* values  = (const float*)d_in[1];  // [N, V]
    const float* queries = (const float*)d_in[2];  // [Q, D]

    char* ws = (char*)d_ws;
    bf16_t* phi = (bf16_t*)ws;                                    // N*D*2 = 4 MB
    bf16_t* plo = (bf16_t*)(ws + (size_t)N_PTS * DIMS * 2);       // 4 MB
    float*  psq = (float*)(ws + (size_t)N_PTS * DIMS * 4);        // 64 KB
    float*  pval = psq + N_PTS;                                   // Q*NSPLIT*4 = 256 KB
    int*    pidx = (int*)(pval + (size_t)NQ * NSPLIT);            // 256 KB

    prep_points_kernel<<<N_PTS, 128, 0, stream>>>(points, phi, plo, psq);

    dim3 grid(NQ / 128, NSPLIT);
    nn_argmin_kernel<<<grid, 256, 0, stream>>>(queries, phi, plo, psq, pval, pidx);

    finalize_kernel<<<NQ, NV, 0, stream>>>(pval, pidx, values, (float*)d_out);
}
